// TransformerEncoderLayer_83683142795381
// MI455X (gfx1250) — matrix operand order repaired and checked by reading, not compiled, not run on hardware
//
#include <hip/hip_runtime.h>
#include <hip/hip_bf16.h>
#include <math.h>

// ---------------------------------------------------------------------------
// TransformerEncoderLayer on gfx1250 (MI455X).
//  - All GEMMs: v_wmma_f32_16x16x32_f16 (f16 operands, f32 accumulate).
//  - GEMM B tiles staged in LDS via global_load_async_to_lds_b128 (ASYNCcnt,
//    double buffered) so the DMA overlaps WMMA issue; 32x64 wave tiles.
//  - All four B fragments materialized before the 8-WMMA burst so ds_loads
//    issue as one clause with partial dscnt waits instead of full stalls.
//  - Attention fused flash-style: [B,H,S,S] scores never touch HBM.
// ---------------------------------------------------------------------------

#define Bsz 4
#define Ssz 2048
#define Dsz 512
#define Hsz 8
#define HDsz 64
#define Fsz 2048
#define BS (Bsz * Ssz)          // 8192 rows

typedef __attribute__((ext_vector_type(16))) _Float16 v16h;
typedef __attribute__((ext_vector_type(8)))  _Float16 v8h;
typedef __attribute__((ext_vector_type(8)))  float    v8f;

// ---- WMMA fragment helpers (wave32, layouts per CDNA5 ISA) -----

// A-matrix 16x32 f16, row-major source with leading dim ld (halfs).
// lane l<16: M=l, K={0..7,16..23}; lane l+16: M=l-16, K={8..15,24..31}.
__device__ __forceinline__ v16h load_a_frag(const _Float16* base, int ld) {
    int lane = threadIdx.x & 31;
    int m  = lane & 15;
    int kb = (lane >> 4) << 3;              // 0 or 8
    const _Float16* p = base + (size_t)m * ld + kb;
    v8h lo = *(const v8h*)(p);
    v8h hi = *(const v8h*)(p + 16);
    v16h a;
#pragma unroll
    for (int i = 0; i < 8; ++i) { a[i] = lo[i]; a[8 + i] = hi[i]; }
    return a;
}

// B-matrix 32x16 f16 from "Bt" storage: N-major rows, K contiguous (ld halfs).
// lanes 0-15: N=lane, K={0..7,16..23}; lanes 16-31: N=lane-16, K={8..15,24..31}.
__device__ __forceinline__ v16h load_b_frag(const _Float16* base, int ld) {
    int lane = threadIdx.x & 31;
    int n  = lane & 15;
    int kb = (lane >> 4) << 3;              // 0 or 8
    const _Float16* p = base + (size_t)n * ld + kb;
    v8h lo = *(const v8h*)(p);
    v8h hi = *(const v8h*)(p + 16);
    v16h b;
#pragma unroll
    for (int i = 0; i < 8; ++i) { b[i] = lo[i]; b[8 + i] = hi[i]; }
    return b;
}

__device__ __forceinline__ v8f wmma_f16(v16h a, v16h b, v8f c) {
    return __builtin_amdgcn_wmma_f32_16x16x32_f16(
        /*neg_a=*/false, a, /*neg_b=*/false, b,
        /*c_mod=*/(short)0, c, /*reuse_a=*/false, /*reuse_b=*/false);
}

// ---- async global->LDS staging (CDNA5 ISA) ----------
// One wavewide instruction moves 32 lanes x 16B = 512B; 8 waves cover a
// 64x32-half (4KB) B tile per stage.  Tracked with ASYNCcnt (per wave,
// in-order completion), so asynccnt<=1 == "older of two stages resident".

__device__ __forceinline__ void async_b128(const _Float16* gsrc, unsigned lds_off) {
    asm volatile("global_load_async_to_lds_b128 %0, %1, off"
                 :: "v"(lds_off),
                    "v"((unsigned long long)(uintptr_t)gsrc)
                 : "memory");
}
__device__ __forceinline__ void wait_async_le1() {
    asm volatile("s_wait_asynccnt 0x1" ::: "memory");
}
__device__ __forceinline__ void wait_async_le0() {
    asm volatile("s_wait_asynccnt 0x0" ::: "memory");
}

// Stage the [64 x 32] B tile (rows n0..n0+63, k-cols kk..kk+31) into LDS.
// tid -> (row = tid>>2, seg = tid&3): one b128 (8 halfs) per thread.
__device__ __forceinline__ void issue_b_stage(const _Float16* Bt, int K, int n0,
                                              int kk, unsigned lds_base) {
    int tid = threadIdx.x;
    int row = tid >> 2, seg = tid & 3;
    const _Float16* src = Bt + (size_t)(n0 + row) * K + kk + seg * 8;
    async_b128(src, lds_base + (unsigned)(tid * 16));
}

// ---- GEMM core: block = 8 waves, tile 256(M) x 64(N), K steps of 32 -------
// Each wave owns 32 rows: 2 A-frags (global) x 4 B-frags (LDS) -> 8 WMMA.
__device__ __forceinline__ void gemm_core_lds(const _Float16* A, const _Float16* Bt,
                                              int K, int m0, int n0,
                                              _Float16* sB /* [2][64][32] */,
                                              v8f acc0[4], v8f acc1[4]) {
    unsigned sb0 = (unsigned)(uintptr_t)(void*)sB;  // LDS aperture: low 32b = offset
    unsigned sb1 = sb0 + 4096;
    int nk = K / 32;
    issue_b_stage(Bt, K, n0, 0, sb0);
    for (int ks = 0; ks < nk; ++ks) {
        const _Float16* sbc = sB + (ks & 1) * 2048;   // 64*32 halfs per buffer
        if (ks + 1 < nk) {
            issue_b_stage(Bt, K, n0, (ks + 1) * 32, (ks & 1) ? sb0 : sb1);
            wait_async_le1();
        } else {
            wait_async_le0();
        }
        __syncthreads();
        const _Float16* ap = A + (size_t)m0 * K + ks * 32;
        __builtin_prefetch(ap + 64, 0, 0);            // global_prefetch_b8
        v16h a0 = load_a_frag(ap, K);
        v16h a1 = load_a_frag(ap + (size_t)16 * K, K);
        // distinct SSA values -> distinct VGPR octets -> ds_loads clause up
        v16h b0 = load_b_frag(sbc + 0 * 512, 32);
        v16h b1 = load_b_frag(sbc + 1 * 512, 32);
        v16h b2 = load_b_frag(sbc + 2 * 512, 32);
        v16h b3 = load_b_frag(sbc + 3 * 512, 32);
        acc0[0] = wmma_f16(a0, b0, acc0[0]);
        acc1[0] = wmma_f16(a1, b0, acc1[0]);
        acc0[1] = wmma_f16(a0, b1, acc0[1]);
        acc1[1] = wmma_f16(a1, b1, acc1[1]);
        acc0[2] = wmma_f16(a0, b2, acc0[2]);
        acc1[2] = wmma_f16(a1, b2, acc1[2]);
        acc0[3] = wmma_f16(a0, b3, acc0[3]);
        acc1[3] = wmma_f16(a1, b3, acc1[3]);
        __syncthreads();                              // buffer reuse fence
    }
}

// ---- pack / convert kernels ----------------------------------------------

__global__ __launch_bounds__(256) void k_f32_to_f16(const float* s, _Float16* d, int n) {
    int i = blockIdx.x * 256 + threadIdx.x;
    if (i < n) d[i] = (_Float16)s[i];
}

// WtQKV[n][d], n = which*512 + h*64 + e, from Wq/Wk/Wv [H,D,HD] f32.
__global__ __launch_bounds__(256) void k_pack_qkv_w(const float* Wq, const float* Wk,
                                                    const float* Wv, _Float16* Wt) {
    int i = blockIdx.x * 256 + threadIdx.x;        // over 1536*512
    if (i >= 1536 * 512) return;
    int n = i >> 9, d = i & 511;
    int which = n >> 9, he = n & 511, h = he >> 6, e = he & 63;
    const float* W = (which == 0) ? Wq : (which == 1) ? Wk : Wv;
    Wt[i] = (_Float16)W[((size_t)h * Dsz + d) * HDsz + e];
}

// dst[n*K + k] = (f16) src[k*N + n]   (transpose + cast)
__global__ __launch_bounds__(256) void k_pack_t(const float* src, _Float16* dst,
                                                int K, int N) {
    int i = blockIdx.x * 256 + threadIdx.x;
    if (i >= K * N) return;
    int n = i / K, k = i % K;
    dst[i] = (_Float16)src[(size_t)k * N + n];
}

// ---- GEMM kernels ---------------------------------------------------------

// QKV projection: xh[BS,512] @ WtQKV -> scatter q[bh,s,e], k[bh,s,e], vT[bh,e,s]
__global__ __launch_bounds__(256) void k_gemm_qkv(const _Float16* xh, const _Float16* Wt,
                                                  const float* bq, const float* bk,
                                                  const float* bv, _Float16* qb,
                                                  _Float16* kb, _Float16* vT) {
    __shared__ __align__(16) _Float16 sB[2 * 64 * 32];
    int wave = threadIdx.x >> 5, lane = threadIdx.x & 31;
    int m0 = blockIdx.x * 256 + wave * 32;
    int n0 = blockIdx.y * 64;
    v8f acc0[4] = {}, acc1[4] = {};
    gemm_core_lds(xh, Wt, Dsz, m0, n0, sB, acc0, acc1);
    int nl = lane & 15, mh = (lane >> 4) * 8;
#pragma unroll
    for (int j = 0; j < 4; ++j) {
        int n = n0 + j * 16 + nl;
        int which = n >> 9, he = n & 511, h = he >> 6, e = he & 63;
        float bias = ((which == 0) ? bq : (which == 1) ? bk : bv)[he];
#pragma unroll
        for (int i = 0; i < 2; ++i) {
            const v8f* acc = i ? acc1 : acc0;
#pragma unroll
            for (int r = 0; r < 8; ++r) {
                int mg = m0 + i * 16 + mh + r;
                int b_ = mg >> 11, s = mg & (Ssz - 1);
                int bh = b_ * Hsz + h;
                _Float16 val = (_Float16)(acc[j][r] + bias);
                if (which == 0)      qb[((size_t)bh * Ssz + s) * HDsz + e] = val;
                else if (which == 1) kb[((size_t)bh * Ssz + s) * HDsz + e] = val;
                else                 vT[((size_t)bh * HDsz + e) * Ssz + s] = val;
            }
        }
    }
}

// O projection: attn[BS,512] @ WoT + bo + x -> res1 (f32)
__global__ __launch_bounds__(256) void k_gemm_o(const _Float16* ah, const _Float16* WoT,
                                                const float* bo, const float* x,
                                                float* res1) {
    __shared__ __align__(16) _Float16 sB[2 * 64 * 32];
    int wave = threadIdx.x >> 5, lane = threadIdx.x & 31;
    int m0 = blockIdx.x * 256 + wave * 32;
    int n0 = blockIdx.y * 64;
    v8f acc0[4] = {}, acc1[4] = {};
    gemm_core_lds(ah, WoT, Dsz, m0, n0, sB, acc0, acc1);
    int nl = lane & 15, mh = (lane >> 4) * 8;
#pragma unroll
    for (int j = 0; j < 4; ++j) {
        int n = n0 + j * 16 + nl;
        float bias = bo[n];
#pragma unroll
        for (int i = 0; i < 2; ++i) {
            const v8f* acc = i ? acc1 : acc0;
#pragma unroll
            for (int r = 0; r < 8; ++r) {
                size_t idx = (size_t)(m0 + i * 16 + mh + r) * Dsz + n;
                res1[idx] = acc[j][r] + bias + x[idx];
            }
        }
    }
}

// FFN1: x1h[BS,512] @ W1T + b1 -> exact GELU -> hbuf f16 [BS,2048]
__global__ __launch_bounds__(256) void k_gemm_ffn1(const _Float16* x1h, const _Float16* W1T,
                                                   const float* b1, _Float16* hbuf) {
    __shared__ __align__(16) _Float16 sB[2 * 64 * 32];
    int wave = threadIdx.x >> 5, lane = threadIdx.x & 31;
    int m0 = blockIdx.x * 256 + wave * 32;
    int n0 = blockIdx.y * 64;
    v8f acc0[4] = {}, acc1[4] = {};
    gemm_core_lds(x1h, W1T, Dsz, m0, n0, sB, acc0, acc1);
    int nl = lane & 15, mh = (lane >> 4) * 8;
#pragma unroll
    for (int j = 0; j < 4; ++j) {
        int n = n0 + j * 16 + nl;
        float bias = b1[n];
#pragma unroll
        for (int i = 0; i < 2; ++i) {
            const v8f* acc = i ? acc1 : acc0;
#pragma unroll
            for (int r = 0; r < 8; ++r) {
                float t = acc[j][r] + bias;
                float g = 0.5f * t * (1.0f + erff(t * 0.70710678118654752f));
                hbuf[(size_t)(m0 + i * 16 + mh + r) * Fsz + n] = (_Float16)g;
            }
        }
    }
}

// FFN2: hbuf[BS,2048] @ W2T + b2 + x1 -> res2 (f32)
__global__ __launch_bounds__(256) void k_gemm_ffn2(const _Float16* hbuf, const _Float16* W2T,
                                                   const float* b2, const float* x1,
                                                   float* res2) {
    __shared__ __align__(16) _Float16 sB[2 * 64 * 32];
    int wave = threadIdx.x >> 5, lane = threadIdx.x & 31;
    int m0 = blockIdx.x * 256 + wave * 32;
    int n0 = blockIdx.y * 64;
    v8f acc0[4] = {}, acc1[4] = {};
    gemm_core_lds(hbuf, W2T, Fsz, m0, n0, sB, acc0, acc1);
    int nl = lane & 15, mh = (lane >> 4) * 8;
#pragma unroll
    for (int j = 0; j < 4; ++j) {
        int n = n0 + j * 16 + nl;
        float bias = b2[n];
#pragma unroll
        for (int i = 0; i < 2; ++i) {
            const v8f* acc = i ? acc1 : acc0;
#pragma unroll
            for (int r = 0; r < 8; ++r) {
                size_t idx = (size_t)(m0 + i * 16 + mh + r) * Dsz + n;
                res2[idx] = acc[j][r] + bias + x1[idx];
            }
        }
    }
}

// ---- flash attention: 1 wave per (b,h, 16-row query tile) -----------------
// q: [BH, S, HD]  k: [BH, S, HD]  vT: [BH, HD, S]  -> attn f16 [B,S,D] concat
__global__ __launch_bounds__(32) void k_attention(const _Float16* qb, const _Float16* kb,
                                                  const _Float16* vT, _Float16* attn) {
    __shared__ float    sc[16][32];
    __shared__ __align__(16) _Float16 pb[16][32];
    __shared__ float    rowf[16];
    int lane = threadIdx.x;
    int blk = blockIdx.x;                      // B*H*(S/16) = 4096
    int mt = blk & 127;
    int bh = blk >> 7;
    int m0 = mt * 16;
    const _Float16* Q  = qb + ((size_t)bh * Ssz + m0) * HDsz;
    const _Float16* Kp = kb + (size_t)bh * Ssz * HDsz;
    const _Float16* Vt = vT + (size_t)bh * HDsz * Ssz;

    v16h aQ0 = load_a_frag(Q + 0, HDsz);
    v16h aQ1 = load_a_frag(Q + 32, HDsz);
    v8f o[4] = {};
    float mi = -3.0e38f, li = 0.0f;
    int nl = lane & 15, mh = (lane >> 4) * 8;

    for (int t0 = 0; t0 < Ssz; t0 += 32) {
        // scores tile [16 x 32] = Q(16x64) . K^T  (K stored [t,e] row-major)
        v8f c0 = {}, c1 = {};
        {
            v16h b00 = load_b_frag(Kp + (size_t)(t0 + 0) * HDsz + 0, HDsz);
            v16h b01 = load_b_frag(Kp + (size_t)(t0 + 0) * HDsz + 32, HDsz);
            v16h b10 = load_b_frag(Kp + (size_t)(t0 + 16) * HDsz + 0, HDsz);
            v16h b11 = load_b_frag(Kp + (size_t)(t0 + 16) * HDsz + 32, HDsz);
            c0 = wmma_f16(aQ0, b00, c0);
            c0 = wmma_f16(aQ1, b01, c0);
            c1 = wmma_f16(aQ0, b10, c1);
            c1 = wmma_f16(aQ1, b11, c1);
        }
#pragma unroll
        for (int r = 0; r < 8; ++r) {
            sc[mh + r][nl]      = c0[r] * 0.125f;   // 1/sqrt(64)
            sc[mh + r][16 + nl] = c1[r] * 0.125f;
        }
        __syncthreads();
        if (lane < 16) {                        // row-owner lanes: online softmax
            float rmax = -3.0e38f;
#pragma unroll
            for (int c = 0; c < 32; ++c) rmax = fmaxf(rmax, sc[lane][c]);
            float mnew = fmaxf(mi, rmax);
            float al = __expf(mi - mnew);
            float ps = 0.0f;
#pragma unroll
            for (int c = 0; c < 32; ++c) {
                float p = __expf(sc[lane][c] - mnew);
                ps += p;
                pb[lane][c] = (_Float16)p;
            }
            li = li * al + ps;
            mi = mnew;
            rowf[lane] = al;
        }
        __syncthreads();
        float av[8];
#pragma unroll
        for (int r = 0; r < 8; ++r) av[r] = rowf[mh + r];
#pragma unroll
        for (int j = 0; j < 4; ++j)
#pragma unroll
            for (int r = 0; r < 8; ++r) o[j][r] *= av[r];
        // O += P(16x32) . V(32x64), V stored transposed [e,t] -> contiguous B frags
        v16h aP = load_a_frag(&pb[0][0], 32);
#pragma unroll
        for (int j = 0; j < 4; ++j) {
            v16h bv_ = load_b_frag(Vt + (size_t)(j * 16) * Ssz + t0, Ssz);
            o[j] = wmma_f16(aP, bv_, o[j]);
        }
        __syncthreads();
    }
    if (lane < 16) rowf[lane] = 1.0f / li;
    __syncthreads();
    int b_ = bh >> 3, h = bh & 7;
#pragma unroll
    for (int j = 0; j < 4; ++j)
#pragma unroll
        for (int r = 0; r < 8; ++r) {
            int s = m0 + mh + r;
            attn[((size_t)(b_ * Ssz + s)) * Dsz + h * HDsz + j * 16 + nl] =
                (_Float16)(o[j][r] * rowf[mh + r]);
        }
}

// ---- LayerNorm: one 256-thread block per row (D=512) ----------------------
__global__ __launch_bounds__(256) void k_layernorm(const float* inp, const float* g,
                                                   const float* b, float* y,
                                                   _Float16* yh) {
    __shared__ float red[256];
    int row = blockIdx.x, t = threadIdx.x;
    const float* r = inp + (size_t)row * Dsz;
    float v0 = r[t], v1 = r[t + 256];
    red[t] = v0 + v1;
    __syncthreads();
    for (int o = 128; o > 0; o >>= 1) {
        if (t < o) red[t] += red[t + o];
        __syncthreads();
    }
    float mu = red[0] * (1.0f / Dsz);
    __syncthreads();
    float d0 = v0 - mu, d1 = v1 - mu;
    red[t] = d0 * d0 + d1 * d1;
    __syncthreads();
    for (int o = 128; o > 0; o >>= 1) {
        if (t < o) red[t] += red[t + o];
        __syncthreads();
    }
    float rs = rsqrtf(red[0] * (1.0f / Dsz) + 1e-12f);
    float o0 = d0 * rs * g[t] + b[t];
    float o1 = d1 * rs * g[t + 256] + b[t + 256];
    if (y) {
        y[(size_t)row * Dsz + t] = o0;
        y[(size_t)row * Dsz + t + 256] = o1;
    }
    if (yh) {
        yh[(size_t)row * Dsz + t] = (_Float16)o0;
        yh[(size_t)row * Dsz + t + 256] = (_Float16)o1;
    }
}

// ---------------------------------------------------------------------------

extern "C" void kernel_launch(void* const* d_in, const int* in_sizes, int n_in,
                              void* d_out, int out_size, void* d_ws, size_t ws_size,
                              hipStream_t stream) {
    const float* x    = (const float*)d_in[0];
    const float* Wq   = (const float*)d_in[1];
    const float* bq   = (const float*)d_in[2];
    const float* Wk   = (const float*)d_in[3];
    const float* bk   = (const float*)d_in[4];
    const float* Wv   = (const float*)d_in[5];
    const float* bv   = (const float*)d_in[6];
    const float* Wo   = (const float*)d_in[7];
    const float* bo   = (const float*)d_in[8];
    const float* W1   = (const float*)d_in[9];
    const float* b1   = (const float*)d_in[10];
    const float* W2   = (const float*)d_in[11];
    const float* b2   = (const float*)d_in[12];
    const float* ln1g = (const float*)d_in[13];
    const float* ln1b = (const float*)d_in[14];
    const float* ln2g = (const float*)d_in[15];
    const float* ln2b = (const float*)d_in[16];
    float* out = (float*)d_out;

    char* w = (char*)d_ws;
    auto carve = [&](size_t bytes) {
        char* p = w;
        w += (bytes + 255) & ~(size_t)255;
        return p;
    };
    _Float16* xh    = (_Float16*)carve((size_t)BS * Dsz * 2);       // x in f16
    _Float16* WtQKV = (_Float16*)carve((size_t)1536 * Dsz * 2);     // [n][d]
    _Float16* WoT   = (_Float16*)carve((size_t)Dsz * Dsz * 2);
    _Float16* W1T   = (_Float16*)carve((size_t)Fsz * Dsz * 2);      // [f][d]
    _Float16* W2T   = (_Float16*)carve((size_t)Dsz * Fsz * 2);      // [d][f]
    _Float16* qbuf  = (_Float16*)carve((size_t)BS * Dsz * 2);       // [BH,S,HD]
    _Float16* kbuf  = (_Float16*)carve((size_t)BS * Dsz * 2);       // [BH,S,HD]
    _Float16* vTb   = (_Float16*)carve((size_t)BS * Dsz * 2);       // [BH,HD,S]
    _Float16* attn  = (_Float16*)carve((size_t)BS * Dsz * 2);       // concat f16
    float*    res   = (float*)   carve((size_t)BS * Dsz * 4);       // res1 then res2
    float*    x1    = (float*)   carve((size_t)BS * Dsz * 4);
    _Float16* x1h   = (_Float16*)carve((size_t)BS * Dsz * 2);
    _Float16* hbuf  = (_Float16*)carve((size_t)BS * Fsz * 2);

    dim3 blk256(256);

    // packs / casts
    k_f32_to_f16<<<dim3((BS * Dsz) / 256), blk256, 0, stream>>>(x, xh, BS * Dsz);
    k_pack_qkv_w<<<dim3((1536 * 512) / 256), blk256, 0, stream>>>(Wq, Wk, Wv, WtQKV);
    k_pack_t<<<dim3((Dsz * Dsz) / 256), blk256, 0, stream>>>(Wo, WoT, Dsz, Dsz);
    k_pack_t<<<dim3((Dsz * Fsz) / 256), blk256, 0, stream>>>(W1, W1T, Dsz, Fsz);
    k_pack_t<<<dim3((Fsz * Dsz) / 256), blk256, 0, stream>>>(W2, W2T, Fsz, Dsz);

    // QKV projection (M=8192, N=1536, K=512)
    k_gemm_qkv<<<dim3(BS / 256, 1536 / 64), blk256, 0, stream>>>(
        xh, WtQKV, bq, bk, bv, qbuf, kbuf, vTb);

    // fused flash attention: 4096 one-wave blocks
    k_attention<<<dim3(Bsz * Hsz * (Ssz / 16)), dim3(32), 0, stream>>>(
        qbuf, kbuf, vTb, attn);

    // O projection + residual
    k_gemm_o<<<dim3(BS / 256, Dsz / 64), blk256, 0, stream>>>(attn, WoT, bo, x, res);
    k_layernorm<<<dim3(BS), blk256, 0, stream>>>(res, ln1g, ln1b, x1, x1h);

    // FFN
    k_gemm_ffn1<<<dim3(BS / 256, Fsz / 64), blk256, 0, stream>>>(x1h, W1T, b1, hbuf);
    k_gemm_ffn2<<<dim3(BS / 256, Dsz / 64), blk256, 0, stream>>>(hbuf, W2T, b2, x1, res);
    k_layernorm<<<dim3(BS), blk256, 0, stream>>>(res, ln2g, ln2b, out, (_Float16*)nullptr);
}
